// RetinaNetInference_41815801594332
// MI455X (gfx1250) — matrix operand order repaired and checked by reading, not compiled, not run on hardware
//
#include <hip/hip_runtime.h>
#include <hip/hip_fp16.h>

#define C_CLS 80
#define KPRE 256
#define MAXC 8192
#define MAXT 100
#define IMG_F 512.0f
#define THR_LOGIT -2.9444389791664403f   // logit(0.05)
#define NEGV (-1.0e9f)
#define NEG_HALF (-5.0e8f)

typedef __attribute__((ext_vector_type(16))) _Float16 v16h;
typedef __attribute__((ext_vector_type(8)))  float    v8f;
typedef __attribute__((ext_vector_type(4)))  unsigned v4u;
typedef __attribute__((ext_vector_type(8)))  unsigned v8u;

__device__ __forceinline__ float iou4(float4 a, float4 b) {
  float areaA = (a.z - a.x) * (a.w - a.y);
  float areaB = (b.z - b.x) * (b.w - b.y);
  float ih = fmaxf(fminf(a.z, b.z) - fmaxf(a.x, b.x), 0.0f);
  float iw = fmaxf(fminf(a.w, b.w) - fmaxf(a.y, b.y), 0.0f);
  float inter = ih * iw;
  return inter / (areaA + areaB - inter + 1e-8f);
}

// ---------------------------------------------------------------------------
// Kernel 1: per-(b,c) threshold-filter + top-256 select + box decode.
// Streams class-c scores via CDNA5 async-to-LDS (ASYNCcnt) double buffering.
// ---------------------------------------------------------------------------
__global__ __launch_bounds__(256) void topk_decode_kernel(
    const float* __restrict__ y_cls, const float* __restrict__ y_bbox,
    const float* __restrict__ anchors, float* __restrict__ topk_score,
    float* __restrict__ topk_box, int A)
{
  __shared__ float s_sc[MAXC];
  __shared__ int   s_ix[MAXC];
  __shared__ int   s_cnt;
  __shared__ float s_stage[2][256];

  const int tid = threadIdx.x;
  const int b = blockIdx.x / C_CLS, c = blockIdx.x % C_CLS;
  if (tid == 0) s_cnt = 0;
  __syncthreads();

  const float* base = y_cls + ((size_t)b * (size_t)A) * C_CLS + c;
  const int nch = (A + 255) >> 8;

  auto issue = [&](int ch, int buf) {
    int a = (ch << 8) + tid;
    if (a < A) {
      unsigned long long gaddr =
          (unsigned long long)(const void*)(base + (size_t)a * C_CLS);
      unsigned laddr = (unsigned)(size_t)&s_stage[buf][tid];
      asm volatile("global_load_async_to_lds_b32 %0, %1, off"
                   :: "v"(laddr), "v"(gaddr) : "memory");
    }
  };

  issue(0, 0);
  for (int ch = 0; ch < nch; ++ch) {
    const int buf = ch & 1;
    if (ch + 1 < nch) {
      issue(ch + 1, buf ^ 1);
      asm volatile("s_wait_asynccnt 1" ::: "memory");  // chunk ch landed
    } else {
      asm volatile("s_wait_asynccnt 0" ::: "memory");
    }
    int a = (ch << 8) + tid;
    if (a < A) {
      float v = s_stage[buf][tid];
      if (v > THR_LOGIT) {
        int pos = atomicAdd(&s_cnt, 1);
        if (pos < MAXC) { s_sc[pos] = v; s_ix[pos] = a; }
      }
    }
  }
  __syncthreads();
  const int cnt = min(s_cnt, MAXC);
  for (int i = tid; i < MAXC; i += 256)
    if (i >= cnt) { s_sc[i] = -3.0e38f; s_ix[i] = 0; }
  __syncthreads();

  // Bitonic sort, descending by logit.
  for (int k = 2; k <= MAXC; k <<= 1) {
    for (int j = k >> 1; j > 0; j >>= 1) {
      for (int t = tid; t < (MAXC >> 1); t += 256) {
        int i = ((t / j) * (j << 1)) + (t % j);
        int p = i + j;
        bool desc = ((i & k) == 0);
        float a0 = s_sc[i], a1 = s_sc[p];
        if (desc ? (a0 < a1) : (a0 > a1)) {
          s_sc[i] = a1; s_sc[p] = a0;
          int x = s_ix[i]; s_ix[i] = s_ix[p]; s_ix[p] = x;
        }
      }
      __syncthreads();
    }
  }

  // Emit top-256 (sigmoid score or NEG) + decoded box per slot.
  const int gbase = blockIdx.x * KPRE;
  float sc = s_sc[tid];
  int   ai = s_ix[tid];
  bool  val = (tid < cnt);
  topk_score[gbase + tid] = val ? 1.0f / (1.0f + __expf(-sc)) : NEGV;

  float ay1 = anchors[ai * 4 + 0], ax1 = anchors[ai * 4 + 1];
  float ay2 = anchors[ai * 4 + 2], ax2 = anchors[ai * 4 + 3];
  float ha = ay2 - ay1, wa = ax2 - ax1;
  float cya = ay1 + 0.5f * ha, cxa = ax1 + 0.5f * wa;
  const float* rel = y_bbox + ((size_t)b * A + ai) * 4;
  float cy = cya + rel[0] * ha, cx = cxa + rel[1] * wa;
  float h = ha * __expf(rel[2]), w = wa * __expf(rel[3]);
  ((float4*)topk_box)[gbase + tid] =
      make_float4(cy - 0.5f * h, cx - 0.5f * w, cy + 0.5f * h, cx + 0.5f * w);
}

// ---------------------------------------------------------------------------
// Kernel 2: per-(b,c) greedy NMS on 256 boxes. One wave32; each lane owns 8
// columns in registers. Rank (cumsum of keep flags) done with two
// v_wmma_f32_16x16x32_f16 ops: P = K*U (row prefixes), F = Lstrict*T + P.
// ---------------------------------------------------------------------------
__global__ __launch_bounds__(32) void nms_class_kernel(
    const float* __restrict__ topk_score, const float* __restrict__ topk_box,
    float* __restrict__ masked)
{
  __shared__ float4   sb[KPRE];
  __shared__ float    ssc[KPRE];
  __shared__ _Float16 kmat[KPRE];
  __shared__ float    tl[16];
  __shared__ float    cums[KPRE];

  const int lane = threadIdx.x;
  const int gbase = blockIdx.x * KPRE;

  for (int t = 0; t < 8; ++t) {
    int j = lane + (t << 5);
    sb[j]  = ((const float4*)topk_box)[gbase + j];
    ssc[j] = topk_score[gbase + j];
  }
  __syncthreads();

  float4 bj[8]; float sj[8]; bool validj[8];
  for (int t = 0; t < 8; ++t) {
    int j = lane + (t << 5);
    bj[t] = sb[j]; sj[t] = ssc[j]; validj[t] = sj[t] > NEG_HALF;
  }

  unsigned sup = 0u;
  for (int i = 0; i < KPRE; ++i) {
    unsigned so = (unsigned)__shfl((int)sup, i & 31, 32);
    bool supi  = (so >> (i >> 5)) & 1u;
    bool kepti = (ssc[i] > NEG_HALF) && !supi;
    if (kepti) {
      float4 bi = sb[i];
      for (int t = 0; t < 8; ++t) {
        int j = lane + (t << 5);
        if (j > i && iou4(bi, bj[t]) > 0.5f) sup |= (1u << t);
      }
    }
  }

  for (int t = 0; t < 8; ++t) {
    int j = lane + (t << 5);
    bool kp = validj[t] && !((sup >> t) & 1u);
    kmat[j] = kp ? (_Float16)1.0f : (_Float16)0.0f;
  }
  __syncthreads();

  // ---- WMMA prefix sum over 256 flags, K[m][n] = keep[16m+n] ----
  const int m = lane & 15, half = lane >> 4;
  const int n = m, grp = half;
  v16h a, bU;
  for (int v = 0; v < 8; ++v) {
    // 16-bit A-matrix layout: lanes0-15 K={2v,2v+1 | 16+..}, lanes16-31 +8.
    int k0 = (v < 4 ? (v << 1) : 16 + ((v - 4) << 1)) + (half << 3);
    a[2*v]   = (k0     < 16) ? kmat[m * 16 + k0]     : (_Float16)0.0f;
    a[2*v+1] = (k0 + 1 < 16) ? kmat[m * 16 + k0 + 1] : (_Float16)0.0f;
    // B layout: same K order as A (rows k0,k0+1), col = n.
    int r0 = k0;
    bU[2*v]   = (r0     < 16 && r0     <= n) ? (_Float16)1.0f : (_Float16)0.0f;
    bU[2*v+1] = (r0 + 1 < 16 && r0 + 1 <= n) ? (_Float16)1.0f : (_Float16)0.0f;
  }
  v8f zero = {};
  v8f P = __builtin_amdgcn_wmma_f32_16x16x32_f16(
      false, a, false, bU, (short)0, zero, false, false);

  // Row totals t[m] = P[m][15] live in lane15 (m=0..7) / lane31 (m=8..15).
  if (lane == 15) { for (int r = 0; r < 8; ++r) tl[r]     = P[r]; }
  if (lane == 31) { for (int r = 0; r < 8; ++r) tl[8 + r] = P[r]; }
  __syncthreads();

  v16h a2, b2;
  for (int v = 0; v < 8; ++v) {
    int k0 = (v < 4 ? (v << 1) : 16 + ((v - 4) << 1)) + (half << 3);
    a2[2*v]   = (k0     < 16 && k0     < m) ? (_Float16)1.0f : (_Float16)0.0f;
    a2[2*v+1] = (k0 + 1 < 16 && k0 + 1 < m) ? (_Float16)1.0f : (_Float16)0.0f;
    int r0 = k0;
    b2[2*v]   = (r0     < 16) ? (_Float16)tl[r0]     : (_Float16)0.0f;
    b2[2*v+1] = (r0 + 1 < 16) ? (_Float16)tl[r0 + 1] : (_Float16)0.0f;
  }
  v8f F = __builtin_amdgcn_wmma_f32_16x16x32_f16(
      false, a2, false, b2, (short)0, P, false, false);

  for (int r = 0; r < 8; ++r) {
    int mm = (half << 3) + r;          // C/D layout: VGPR r -> M=r (+8)
    cums[mm * 16 + n] = F[r];
  }
  __syncthreads();

  for (int t = 0; t < 8; ++t) {
    int j = lane + (t << 5);
    bool kp = validj[t] && !((sup >> t) & 1u) &&
              (cums[j] <= (float)MAXT + 0.5f);   // rank = cumsum-1 < 100
    masked[gbase + j] = kp ? sj[t] : NEGV;
  }
}

// ---------------------------------------------------------------------------
// Kernel 3: per-batch top-100, rescale/clip, second NMS (0.7), stable compact.
// The 80 KB masked-score slab is staged into LDS with one TDM descriptor
// (tensor_load_to_lds, TENSORcnt-tracked) issued by wave 0.
// ---------------------------------------------------------------------------
__global__ __launch_bounds__(256) void final_kernel(
    const float* __restrict__ masked, const float* __restrict__ topk_box,
    const int* __restrict__ orig_hs, const int* __restrict__ orig_ws,
    float* __restrict__ out_boxes, float* __restrict__ out_scores,
    float* __restrict__ out_cls, float* __restrict__ out_nv)
{
  __shared__ float  lsc[C_CLS * KPRE];   // 80 KB, fine in 320 KB LDS
  __shared__ float  red_s[256];
  __shared__ int    red_i[256];
  __shared__ float  sel_s[MAXT];
  __shared__ int    sel_i[MAXT];
  __shared__ int    s_count;
  __shared__ float4 rbx[MAXT];
  __shared__ float  rsc[MAXT], rcl[MAXT];
  __shared__ int    ord[MAXT];
  __shared__ int    s_nv;

  const int tid = threadIdx.x;
  const int b = blockIdx.x;
  const int Ntot = C_CLS * KPRE;
  const size_t mb = (size_t)b * Ntot;

  // --- TDM: one descriptor copies masked[b, 0:20480] (1-D tile) into LDS ---
  if (tid < 32) {
    unsigned long long gaddr =
        (unsigned long long)(const void*)(masked + mb);
    unsigned laddr = (unsigned)(size_t)&lsc[0];
    v4u g0;
    g0[0] = 1u;                                       // count=1, user D#
    g0[1] = laddr;                                    // lds_addr
    g0[2] = (unsigned)(gaddr & 0xFFFFFFFFull);        // global_addr[31:0]
    g0[3] = (unsigned)((gaddr >> 32) & 0x01FFFFFFull) // global_addr[56:32]
            | (2u << 30);                             // type = 2 ("image")
    v8u g1;
    g1[0] = 0x00020000u;                    // data_size=2 (4B), no multicast
    g1[1] = ((unsigned)Ntot & 0xFFFFu) << 16;   // tensor_dim0[15:0]
    g1[2] = ((unsigned)Ntot >> 16) | (1u << 16);// tensor_dim0[31:16]|dim1=1
    g1[3] = ((unsigned)Ntot & 0xFFFFu) << 16;   // tile_dim0 = 20480
    g1[4] = 0u;                                 // tile_dim1/2 unused
    g1[5] = (unsigned)Ntot;                     // tensor_dim0_stride
    g1[6] = 0u;
    g1[7] = 0u;
    asm volatile("tensor_load_to_lds %0, %1" :: "s"(g0), "s"(g1) : "memory");
    __builtin_amdgcn_s_wait_tensorcnt(0);
  }
  __syncthreads();

  for (int it = 0; it < MAXT; ++it) {
    float best = -3.0e38f; int bi = 0;
    for (int i = tid; i < Ntot; i += 256) {
      float v = lsc[i];
      if (v > best) { best = v; bi = i; }
    }
    red_s[tid] = best; red_i[tid] = bi;
    __syncthreads();
    for (int off = 128; off > 0; off >>= 1) {
      if (tid < off) {
        float o = red_s[tid + off];
        if (o > red_s[tid] || (o == red_s[tid] && red_i[tid + off] < red_i[tid])) {
          red_s[tid] = o; red_i[tid] = red_i[tid + off];
        }
      }
      __syncthreads();
    }
    if (tid == 0) {
      sel_s[it] = red_s[0]; sel_i[it] = red_i[0];
      lsc[red_i[0]] = -3.0e38f;
    }
    __syncthreads();
  }
  if (tid == 0) {
    int cnt = 0;
    for (int i = 0; i < MAXT; ++i) if (sel_s[i] > NEG_HALF) ++cnt;
    s_count = cnt;
  }
  __syncthreads();

  const int count = s_count;
  const float hh = (float)orig_hs[b], ww = (float)orig_ws[b];
  const float rh = hh / IMG_F, rw = ww / IMG_F;
  if (tid < MAXT) {
    bool sv = tid < count;
    int slot = sel_i[tid];
    float4 bx = ((const float4*)topk_box)[mb + slot];
    if (!sv) bx = make_float4(0.f, 0.f, 0.f, 0.f);
    float4 rb;
    rb.x = fminf(fmaxf(bx.x * rh, 0.f), hh);
    rb.y = fminf(fmaxf(bx.y * rw, 0.f), ww);
    rb.z = fminf(fmaxf(bx.z * rh, 0.f), hh);
    rb.w = fminf(fmaxf(bx.w * rw, 0.f), ww);
    rbx[tid] = rb;
    rsc[tid] = sv ? sel_s[tid] : 0.0f;
    rcl[tid] = sv ? (float)(slot / KPRE) : 0.0f;
  }
  __syncthreads();

  if (tid < 32) {
    const int lane = tid;
    unsigned sup2 = 0u;
    float4 bj[4];
    for (int t = 0; t < 4; ++t) {
      int j = lane + (t << 5);
      bj[t] = (j < MAXT) ? rbx[j] : make_float4(0.f, 0.f, 0.f, 0.f);
    }
    for (int i = 0; i < MAXT; ++i) {
      unsigned so = (unsigned)__shfl((int)sup2, i & 31, 32);
      bool supi  = (so >> (i >> 5)) & 1u;
      bool kepti = (i < count) && !supi;
      if (kepti) {
        float4 bi = rbx[i];
        for (int t = 0; t < 4; ++t) {
          int j = lane + (t << 5);
          if (j < MAXT && j > i && iou4(bi, bj[t]) > 0.7f) sup2 |= (1u << t);
        }
      }
    }
    unsigned long long msk[4]; int tot[4];
    for (int t = 0; t < 4; ++t) {
      int j = lane + (t << 5);
      bool kp = (j < MAXT) && (j < count) && !((sup2 >> t) & 1u);
      msk[t] = __ballot(kp ? 1 : 0);
      tot[t] = __popcll(msk[t]);
    }
    for (int t = 0; t < 4; ++t) {
      int j = lane + (t << 5);
      bool kp = (j < MAXT) && (j < count) && !((sup2 >> t) & 1u);
      if (kp) {
        int rank = 0;
        for (int u = 0; u < t; ++u) rank += tot[u];
        rank += __popcll(msk[t] & ((1ull << lane) - 1ull));
        ord[rank] = j;
      }
    }
    if (lane == 0) s_nv = tot[0] + tot[1] + tot[2] + tot[3];
  }
  __syncthreads();

  const int nv = s_nv;
  if (tid < MAXT) {
    float4 ob = make_float4(0.f, 0.f, 0.f, 0.f);
    float os = 0.f, oc = 0.f;
    if (tid < nv) { int j = ord[tid]; ob = rbx[j]; os = rsc[j]; oc = rcl[j]; }
    ((float4*)out_boxes)[(size_t)b * MAXT + tid] = ob;
    out_scores[(size_t)b * MAXT + tid] = os;
    out_cls[(size_t)b * MAXT + tid]    = oc;
  }
  if (tid == 0) out_nv[b] = (float)nv;
}

// ---------------------------------------------------------------------------
extern "C" void kernel_launch(void* const* d_in, const int* in_sizes, int n_in,
                              void* d_out, int out_size, void* d_ws, size_t ws_size,
                              hipStream_t stream) {
  (void)n_in; (void)out_size; (void)ws_size;
  const float* y_cls   = (const float*)d_in[0];
  const float* y_bbox  = (const float*)d_in[1];
  const float* anchors = (const float*)d_in[2];
  const int*   orig_hs = (const int*)d_in[3];
  const int*   orig_ws = (const int*)d_in[4];
  const int A  = in_sizes[2] / 4;
  const int Bn = in_sizes[3];

  // Workspace layout (floats): topk scores | topk boxes | masked scores
  float* topk_score = (float*)d_ws;
  float* topk_box   = topk_score + (size_t)Bn * C_CLS * KPRE;
  float* maskedp    = topk_box   + (size_t)Bn * C_CLS * KPRE * 4;

  float* out        = (float*)d_out;
  float* out_boxes  = out;
  float* out_scores = out + (size_t)Bn * MAXT * 4;
  float* out_cls    = out_scores + (size_t)Bn * MAXT;
  float* out_nv     = out_cls + (size_t)Bn * MAXT;

  topk_decode_kernel<<<Bn * C_CLS, 256, 0, stream>>>(
      y_cls, y_bbox, anchors, topk_score, topk_box, A);
  nms_class_kernel<<<Bn * C_CLS, 32, 0, stream>>>(
      topk_score, topk_box, maskedp);
  final_kernel<<<Bn, 256, 0, stream>>>(
      maskedp, topk_box, orig_hs, orig_ws,
      out_boxes, out_scores, out_cls, out_nv);
}
